// SelfAttentionBase_60481729462550
// MI455X (gfx1250) — matrix operand order repaired and checked by reading, not compiled, not run on hardware
//
#include <hip/hip_runtime.h>
#include <hip/hip_bf16.h>

// ---------------------------------------------------------------------------
// Self-attention for MI455X (gfx1250, wave32, WMMA).
//   B=2, S=2048, D=1024, H=16, HD=64.  All matmuls via v_wmma_f32_16x16x32_f16.
// ---------------------------------------------------------------------------

typedef __attribute__((ext_vector_type(16))) _Float16 v16h;
typedef __attribute__((ext_vector_type(8)))  _Float16 v8h;
typedef __attribute__((ext_vector_type(8)))  float    v8f;
typedef __attribute__((ext_vector_type(4)))  int      v4i;

#define B_  2
#define S_  2048
#define D_  1024
#define H_  16
#define HD_ 64
#define M_  (B_ * S_)   // 4096 total rows

// CDNA5 async global->LDS copy (ASYNCcnt path), guarded so we always compile.
#if defined(__has_builtin)
#if __has_builtin(__builtin_amdgcn_global_load_async_to_lds_b128) && \
    __has_builtin(__builtin_amdgcn_s_wait_asynccnt)
#define HAVE_ASYNC_LDS 1
#endif
#endif

#ifdef HAVE_ASYNC_LDS
typedef __attribute__((address_space(1))) v4i* as1_b128;
typedef __attribute__((address_space(3))) v4i* as3_b128;
#endif

__device__ __forceinline__ v8f wmma_f16(v16h a, v16h b, v8f c) {
  // (neg_a, A, neg_b, B, c_mod, C, reuse_a, reuse_b)
  return __builtin_amdgcn_wmma_f32_16x16x32_f16(false, a, false, b, (short)0, c,
                                                false, false);
}

// XOR-swizzle reductions across the 16-lane group (masks < 16 stay in-half).
// group-of-32 pattern: offset = and_mask | (or<<5) | (xor<<10)
__device__ __forceinline__ float red_max16(float v) {
  v = fmaxf(v, __int_as_float(__builtin_amdgcn_ds_swizzle(__float_as_int(v), 0x041f)));
  v = fmaxf(v, __int_as_float(__builtin_amdgcn_ds_swizzle(__float_as_int(v), 0x081f)));
  v = fmaxf(v, __int_as_float(__builtin_amdgcn_ds_swizzle(__float_as_int(v), 0x101f)));
  v = fmaxf(v, __int_as_float(__builtin_amdgcn_ds_swizzle(__float_as_int(v), 0x201f)));
  return v;
}
__device__ __forceinline__ float red_sum16(float v) {
  v += __int_as_float(__builtin_amdgcn_ds_swizzle(__float_as_int(v), 0x041f));
  v += __int_as_float(__builtin_amdgcn_ds_swizzle(__float_as_int(v), 0x081f));
  v += __int_as_float(__builtin_amdgcn_ds_swizzle(__float_as_int(v), 0x101f));
  v += __int_as_float(__builtin_amdgcn_ds_swizzle(__float_as_int(v), 0x201f));
  return v;
}

__device__ __forceinline__ v16h cat16(v8h lo, v8h hi) {
  return __builtin_shufflevector(lo, hi, 0, 1, 2, 3, 4, 5, 6, 7,
                                         8, 9, 10, 11, 12, 13, 14, 15);
}

// ---------------------------------------------------------------------------
// fp32 -> f16 elementwise convert (weights)
// ---------------------------------------------------------------------------
__global__ void cvt_f16(const float* __restrict__ s, _Float16* __restrict__ d, int n) {
  for (int i = blockIdx.x * blockDim.x + threadIdx.x; i < n;
       i += gridDim.x * blockDim.x)
    d[i] = (_Float16)s[i];
}

// ---------------------------------------------------------------------------
// Tiled GEMM:  D[M,N] = A[M,K] * Bw[K,N] + bias[N]
//   A is fp32 (converted on load) or f16 (async-copied to LDS when available).
//   WG = 256 threads = 8 waves; WG tile 128x128; wave tile 64x32 (4x2 WMMA).
// ---------------------------------------------------------------------------
template <bool A_F32, bool OUT_F16>
__global__ __launch_bounds__(256) void gemm_tile(
    const void* __restrict__ Aptr, const _Float16* __restrict__ Bw,
    const float* __restrict__ bias, void* __restrict__ Dptr,
    int M, int N, int K) {
  __shared__ _Float16 ldsA[128][40];  // [m][k]   (pad 40 halves = 80B rows)
  __shared__ _Float16 ldsB[128][40];  // [n][k]   (transposed)

  const int tid  = threadIdx.x;
  const int lane = tid & 31;
  const int wv   = tid >> 5;
  const int wrow = wv >> 2;       // 0..1
  const int wcol = wv & 3;        // 0..3
  const int half = lane >> 4;     // 0/1
  const int mr   = lane & 15;
  const int rowBase = blockIdx.x * 128;
  const int nBase   = blockIdx.y * 128;

  v8f acc[4][2];
#pragma unroll
  for (int rt = 0; rt < 4; ++rt)
#pragma unroll
    for (int ct = 0; ct < 2; ++ct) acc[rt][ct] = (v8f)0.0f;

  for (int kk = 0; kk < K; kk += 32) {
    // ---- stage A tile [128 x 32] into LDS ----
    if (A_F32) {
      const float* A = (const float*)Aptr;
#pragma unroll
      for (int i = 0; i < 4; ++i) {
        int c = tid + i * 256;     // 1024 chunks of float4
        int r = c >> 3, cf = c & 7;
        float4 f = *reinterpret_cast<const float4*>(
            A + (size_t)(rowBase + r) * K + kk + cf * 4);
        _Float16* dst = &ldsA[r][cf * 4];
        dst[0] = (_Float16)f.x; dst[1] = (_Float16)f.y;
        dst[2] = (_Float16)f.z; dst[3] = (_Float16)f.w;
      }
    } else {
      const _Float16* A = (const _Float16*)Aptr;
#pragma unroll
      for (int i = 0; i < 2; ++i) {
        int c = tid + i * 256;     // 512 chunks of 8 halves
        int r = c >> 2, c8 = c & 3;
        const _Float16* src = A + (size_t)(rowBase + r) * K + kk + c8 * 8;
#ifdef HAVE_ASYNC_LDS
        __builtin_amdgcn_global_load_async_to_lds_b128(
            (as1_b128)src, (as3_b128)&ldsA[r][c8 * 8], 0, 0);
#else
        *reinterpret_cast<v8h*>(&ldsA[r][c8 * 8]) =
            *reinterpret_cast<const v8h*>(src);
#endif
      }
    }
    // ---- stage B tile [32 x 128] transposed into LDS ----
#pragma unroll
    for (int i = 0; i < 2; ++i) {
      int c  = tid + i * 256;      // 512 chunks of 8 halves along n
      int kr = c >> 4, cn = c & 15;
      v8h v = *reinterpret_cast<const v8h*>(
          Bw + (size_t)(kk + kr) * N + nBase + cn * 8);
#pragma unroll
      for (int j = 0; j < 8; ++j) ldsB[cn * 8 + j][kr] = v[j];
    }
    if (kk + 32 < K) {  // hint next B tile toward the caches
      __builtin_prefetch((const void*)(Bw + (size_t)(kk + 32) * N + nBase), 0, 0);
    }
#ifdef HAVE_ASYNC_LDS
    if (!A_F32) __builtin_amdgcn_s_wait_asynccnt(0);
#endif
    __syncthreads();

    // ---- B fragments for this wave's two 16-col tiles ----
    v16h bf[2];
#pragma unroll
    for (int ct = 0; ct < 2; ++ct) {
      const _Float16* bp = &ldsB[wcol * 32 + ct * 16 + mr][half * 16];
      bf[ct] = cat16(*reinterpret_cast<const v8h*>(bp),
                     *reinterpret_cast<const v8h*>(bp + 8));
    }
    // ---- A fragments + WMMA ----
#pragma unroll
    for (int rt = 0; rt < 4; ++rt) {
      const _Float16* ap = &ldsA[wrow * 64 + rt * 16 + mr][0];
      v16h af = cat16(*reinterpret_cast<const v8h*>(ap + half * 16),
                      *reinterpret_cast<const v8h*>(ap + half * 16 + 8));
      acc[rt][0] = wmma_f16(af, bf[0], acc[rt][0]);
      acc[rt][1] = wmma_f16(af, bf[1], acc[rt][1]);
    }
    __syncthreads();
  }

  // ---- epilogue: D layout m = vgpr + 8*half, n = lane%16 ----
#pragma unroll
  for (int ct = 0; ct < 2; ++ct) {
    int n = nBase + wcol * 32 + ct * 16 + mr;
    float bvv = bias[n];
#pragma unroll
    for (int rt = 0; rt < 4; ++rt) {
#pragma unroll
      for (int vv = 0; vv < 8; ++vv) {
        int m = rowBase + wrow * 64 + rt * 16 + vv + half * 8;
        float val = acc[rt][ct][vv] + bvv;
        if (OUT_F16)
          ((_Float16*)Dptr)[(size_t)m * N + n] = (_Float16)val;
        else
          ((float*)Dptr)[(size_t)m * N + n] = val;
      }
    }
  }
}

// ---------------------------------------------------------------------------
// Flash attention over f16 projected Q/K/V ([M, D] row-major, head = column
// group h*64..h*64+63).  One WG (4 waves) per (b, h, 128-query block); each
// wave owns 32 query rows.  Online softmax, scores never hit HBM.
// ---------------------------------------------------------------------------
__global__ __launch_bounds__(128) void flash_attn(
    const _Float16* __restrict__ Qp, const _Float16* __restrict__ Kp,
    const _Float16* __restrict__ Vp, _Float16* __restrict__ Ctx) {
  __shared__ _Float16 ldsV[64][72];       // [feat][key]  (transposed V block)
  __shared__ _Float16 ldsP[4][32][72];    // per-wave P   [qrow][key]

  const int tid  = threadIdx.x;
  const int lane = tid & 31;
  const int wv   = tid >> 5;
  const int half = lane >> 4;
  const int mr   = lane & 15;
  const int b    = blockIdx.y >> 4;       // H_ = 16
  const int h    = blockIdx.y & 15;
  const size_t rowB = (size_t)b * S_;
  const int qBase = blockIdx.x * 128 + wv * 32;

  // ---- Q fragments (held for whole kernel): A-matrix 16x32 layout ----
  v16h qa[2][2];
#pragma unroll
  for (int rt = 0; rt < 2; ++rt)
#pragma unroll
    for (int kq = 0; kq < 2; ++kq) {
      const _Float16* qp =
          Qp + (rowB + qBase + rt * 16 + mr) * D_ + h * HD_ + kq * 32;
      qa[rt][kq] = cat16(*reinterpret_cast<const v8h*>(qp + half * 16),
                         *reinterpret_cast<const v8h*>(qp + half * 16 + 8));
    }

  v8f oacc[2][4];
  float mi[2][8], li[2][8];
#pragma unroll
  for (int rt = 0; rt < 2; ++rt) {
#pragma unroll
    for (int ft = 0; ft < 4; ++ft) oacc[rt][ft] = (v8f)0.0f;
#pragma unroll
    for (int vv = 0; vv < 8; ++vv) { mi[rt][vv] = -1e30f; li[rt][vv] = 0.0f; }
  }

  for (int j = 0; j < S_; j += 64) {
    __syncthreads();  // everyone done reading previous V block
    // ---- cooperative load of V block [64 keys x 64 feats], transposed ----
#pragma unroll
    for (int i = 0; i < 4; ++i) {
      int c  = tid + i * 128;     // 512 chunks of 8 halves
      int kr = c >> 3, cf = c & 7;
      v8h v = *reinterpret_cast<const v8h*>(
          Vp + (rowB + j + kr) * D_ + h * HD_ + cf * 8);
#pragma unroll
      for (int jj = 0; jj < 8; ++jj) ldsV[cf * 8 + jj][kr] = v[jj];
    }
    __syncthreads();

    // ---- scores S = (Q K^T) * 1/sqrt(HD): 32 rows x 64 keys per wave ----
    v8f sacc[2][4];
#pragma unroll
    for (int kt = 0; kt < 4; ++kt) {
      v16h bk[2];
#pragma unroll
      for (int kq = 0; kq < 2; ++kq) {
        const _Float16* kp = Kp + (rowB + j + kt * 16 + mr) * D_ + h * HD_ +
                             kq * 32 + half * 16;
        bk[kq] = cat16(*reinterpret_cast<const v8h*>(kp),
                       *reinterpret_cast<const v8h*>(kp + 8));
      }
#pragma unroll
      for (int rt = 0; rt < 2; ++rt) {
        sacc[rt][kt] = wmma_f16(qa[rt][0], bk[0], (v8f)0.0f);
        sacc[rt][kt] = wmma_f16(qa[rt][1], bk[1], sacc[rt][kt]);
      }
    }

    // ---- online softmax update (rows live across 16-lane groups) ----
#pragma unroll
    for (int rt = 0; rt < 2; ++rt) {
#pragma unroll
      for (int vv = 0; vv < 8; ++vv) {
        float mx = -1e30f;
#pragma unroll
        for (int kt = 0; kt < 4; ++kt) {
          sacc[rt][kt][vv] *= 0.125f;  // 1/sqrt(64)
          mx = fmaxf(mx, sacc[rt][kt][vv]);
        }
        mx = red_max16(mx);
        float mnew = fmaxf(mi[rt][vv], mx);
        float f    = __expf(mi[rt][vv] - mnew);
        mi[rt][vv] = mnew;
        float sum = 0.0f;
#pragma unroll
        for (int kt = 0; kt < 4; ++kt) {
          float p = __expf(sacc[rt][kt][vv] - mnew);
          sacc[rt][kt][vv] = p;
          sum += p;
        }
        sum = red_sum16(sum);
        li[rt][vv] = li[rt][vv] * f + sum;
#pragma unroll
        for (int ft = 0; ft < 4; ++ft) oacc[rt][ft][vv] *= f;
      }
    }

    // ---- redistribute P (D-layout) -> A-layout via per-wave LDS buffer ----
#pragma unroll
    for (int rt = 0; rt < 2; ++rt)
#pragma unroll
      for (int kt = 0; kt < 4; ++kt)
#pragma unroll
        for (int vv = 0; vv < 8; ++vv)
          ldsP[wv][rt * 16 + vv + half * 8][kt * 16 + mr] =
              (_Float16)sacc[rt][kt][vv];
    // same-wave LDS ops are in-order; compiler inserts s_wait_dscnt

    // ---- O += P @ V ----
#pragma unroll
    for (int kp = 0; kp < 2; ++kp) {  // 64 keys = 2 k-steps of 32
      v16h pa[2];
#pragma unroll
      for (int rt = 0; rt < 2; ++rt) {
        const _Float16* pp = &ldsP[wv][rt * 16 + mr][kp * 32];
        pa[rt] = cat16(*reinterpret_cast<const v8h*>(pp + half * 16),
                       *reinterpret_cast<const v8h*>(pp + half * 16 + 8));
      }
#pragma unroll
      for (int ft = 0; ft < 4; ++ft) {
        const _Float16* vp = &ldsV[ft * 16 + mr][kp * 32 + half * 16];
        v16h bv = cat16(*reinterpret_cast<const v8h*>(vp),
                        *reinterpret_cast<const v8h*>(vp + 8));
#pragma unroll
        for (int rt = 0; rt < 2; ++rt)
          oacc[rt][ft] = wmma_f16(pa[rt], bv, oacc[rt][ft]);
      }
    }
  }

  // ---- normalize and write context (f16, [M, D] with head columns) ----
#pragma unroll
  for (int rt = 0; rt < 2; ++rt) {
    float inv[8];
#pragma unroll
    for (int vv = 0; vv < 8; ++vv) inv[vv] = 1.0f / li[rt][vv];
#pragma unroll
    for (int ft = 0; ft < 4; ++ft) {
      int col = h * HD_ + ft * 16 + mr;
#pragma unroll
      for (int vv = 0; vv < 8; ++vv) {
        size_t row = rowB + qBase + rt * 16 + vv + half * 8;
        Ctx[row * D_ + col] = (_Float16)(oacc[rt][ft][vv] * inv[vv]);
      }
    }
  }
}

// ---------------------------------------------------------------------------
// Host launcher
// ---------------------------------------------------------------------------
extern "C" void kernel_launch(void* const* d_in, const int* in_sizes, int n_in,
                              void* d_out, int out_size, void* d_ws,
                              size_t ws_size, hipStream_t stream) {
  (void)in_sizes; (void)n_in; (void)out_size; (void)ws_size;
  const float* q  = (const float*)d_in[0];
  const float* k  = (const float*)d_in[1];
  const float* v  = (const float*)d_in[2];
  const float* Wq = (const float*)d_in[3];
  const float* bq = (const float*)d_in[4];
  const float* Wk = (const float*)d_in[5];
  const float* bk = (const float*)d_in[6];
  const float* Wv = (const float*)d_in[7];
  const float* bv = (const float*)d_in[8];
  const float* Wo = (const float*)d_in[9];
  const float* bo = (const float*)d_in[10];

  // workspace layout (bytes):
  //   4 x f16 weights (2 MB each) | Qp | Kp | Vp | Ctx  (f16 [M,D], 8 MB each)
  char* ws = (char*)d_ws;
  const size_t WSZ = (size_t)D_ * D_ * sizeof(_Float16);  // 2 MB
  const size_t PSZ = (size_t)M_ * D_ * sizeof(_Float16);  // 8 MB
  _Float16* Wq16 = (_Float16*)(ws + 0 * WSZ);
  _Float16* Wk16 = (_Float16*)(ws + 1 * WSZ);
  _Float16* Wv16 = (_Float16*)(ws + 2 * WSZ);
  _Float16* Wo16 = (_Float16*)(ws + 3 * WSZ);
  _Float16* Qp   = (_Float16*)(ws + 4 * WSZ);
  _Float16* Kp   = (_Float16*)(ws + 4 * WSZ + 1 * PSZ);
  _Float16* Vp   = (_Float16*)(ws + 4 * WSZ + 2 * PSZ);
  _Float16* Ctx  = (_Float16*)(ws + 4 * WSZ + 3 * PSZ);

  // 1) weights -> f16
  cvt_f16<<<dim3(2048), 256, 0, stream>>>(Wq, Wq16, D_ * D_);
  cvt_f16<<<dim3(2048), 256, 0, stream>>>(Wk, Wk16, D_ * D_);
  cvt_f16<<<dim3(2048), 256, 0, stream>>>(Wv, Wv16, D_ * D_);
  cvt_f16<<<dim3(2048), 256, 0, stream>>>(Wo, Wo16, D_ * D_);

  // 2) Q/K/V projections (fp32 A converted on load, f16 out)
  dim3 gGemm(M_ / 128, D_ / 128);
  gemm_tile<true, true><<<gGemm, 256, 0, stream>>>(q, Wq16, bq, Qp, M_, D_, D_);
  gemm_tile<true, true><<<gGemm, 256, 0, stream>>>(k, Wk16, bk, Kp, M_, D_, D_);
  gemm_tile<true, true><<<gGemm, 256, 0, stream>>>(v, Wv16, bv, Vp, M_, D_, D_);

  // 3) flash attention -> f16 context
  flash_attn<<<dim3(S_ / 128, B_ * H_), 128, 0, stream>>>(Qp, Kp, Vp, Ctx);

  // 4) output projection (f16 A, fp32 out)
  gemm_tile<false, false><<<gGemm, 256, 0, stream>>>(Ctx, Wo16, bo, d_out,
                                                     M_, D_, D_);
}
